// PerformerLayer_54030688584105
// MI455X (gfx1250) — matrix operand order repaired and checked by reading, not compiled, not run on hardware
//
#include <hip/hip_runtime.h>
#include <hip/hip_bf16.h>

// ---------------- constants ----------------
constexpr int Bc = 4, Lc = 4096, Dc = 1024, Hc = 16, Mc = 256, HDc = 64;
constexpr int Nc = Bc * Lc;          // 16384 rows
constexpr int D4 = 4 * Dc;           // 4096

typedef __attribute__((ext_vector_type(16))) __bf16 bf16x16;
typedef __attribute__((ext_vector_type(8)))  float  f32x8;
typedef __attribute__((ext_vector_type(4)))  unsigned v4u;
typedef __attribute__((ext_vector_type(8)))  int      v8i;
typedef __attribute__((ext_vector_type(4)))  int      v4i;

union ABFrag { bf16x16 v; uint4 q[2]; };

static __device__ __forceinline__ __bf16 f2bf(float f) { return (__bf16)f; }
static __device__ __forceinline__ float  bf2f(__bf16 h) { return (float)h; }
static __device__ __forceinline__ unsigned pk2(float a, float b) {
    union { __bf16 h[2]; unsigned u; } o;
    o.h[0] = (__bf16)a; o.h[1] = (__bf16)b;
    return o.u;
}
static __device__ __forceinline__ void st2(__bf16* p0, __bf16* p1, float a, float b) {
    *p0 = (__bf16)a; *p1 = (__bf16)b;
}

// Async global->LDS copy of 16 bytes (per lane), tracked by ASYNCcnt.
static __device__ __forceinline__ void async_b128(const __bf16* gaddr, void* lds) {
    asm volatile("global_load_async_to_lds_b128 %0, %1, off"
                 :: "v"((unsigned)(size_t)lds),
                    "v"((unsigned long long)(size_t)gaddr)
                 : "memory");
}
static __device__ __forceinline__ void wait_async() {
    asm volatile("s_wait_asynccnt 0x0" ::: "memory");
}

// ---- Tensor Data Mover: 2D tile load, 2-byte elements, rows padded 64B->80B ----
// Loads tile [tile1 rows x tile0 cols] of a row-major [dim1 x dim0] bf16 tensor
// (row stride stride0 elements) into LDS at ldsOff with 16B padding after each
// 64B row (pad_interval code 3 = 64B, pad_amount code 3 = 4 DWORDs), matching
// a [rows][40] __bf16 LDS layout for tile0 == 32.
static __device__ __forceinline__ void tdm_load_tile_2d(const void* gtile, unsigned ldsOff,
                                                        unsigned tile0, unsigned tile1,
                                                        unsigned dim0, unsigned dim1,
                                                        unsigned long long stride0) {
#if __has_builtin(__builtin_amdgcn_tensor_load_to_lds)
    unsigned long long ga = (unsigned long long)(size_t)gtile;
    v4u g0;
    g0.x = 1u;                                              // count=1, is_restore=0, gather off
    g0.y = ldsOff;                                          // lds_addr
    g0.z = (unsigned)(ga & 0xFFFFFFFFu);                    // global_addr[31:0]
    g0.w = (unsigned)((ga >> 32) & 0x01FFFFFFu) | (2u << 30); // global_addr[56:32], type=2
    v8i g1;
    g1[0] = (int)((1u << 16) | (1u << 20) | (3u << 22) | (3u << 25)); // data_size=2B, pad on
    g1[1] = (int)((dim0 & 0xFFFFu) << 16);                  // tensor_dim0[15:0]
    g1[2] = (int)((dim0 >> 16) | ((dim1 & 0xFFFFu) << 16)); // tensor_dim0[31:16], dim1[15:0]
    g1[3] = (int)((dim1 >> 16) | ((tile0 & 0xFFFFu) << 16));// tensor_dim1[31:16], tile_dim0
    g1[4] = (int)(tile1 & 0xFFFFu);                         // tile_dim1, tile_dim2=0
    g1[5] = (int)(unsigned)(stride0 & 0xFFFFFFFFu);         // tensor_dim0_stride[31:0]
    g1[6] = (int)(unsigned)((stride0 >> 32) & 0xFFFFu);     // stride[47:32], dim1_stride lo=0
    g1[7] = 0;
    v4i zz4 = {0, 0, 0, 0};
    v8i zz8 = {0, 0, 0, 0, 0, 0, 0, 0};
    __builtin_amdgcn_tensor_load_to_lds(g0, g1, zz4, zz4, zz8, 0);
#else
    (void)gtile; (void)ldsOff; (void)tile0; (void)tile1; (void)dim0; (void)dim1; (void)stride0;
#endif
}

// ---------------- f32 -> bf16 convert ----------------
__global__ __launch_bounds__(256) void cvt_f32_bf16(const float* __restrict__ in,
                                                    __bf16* __restrict__ out, long n4) {
    long i = (long)blockIdx.x * 256 + threadIdx.x;
    if (i < n4) {
        float4 f = ((const float4*)in)[i];
        uint2 o; o.x = pk2(f.x, f.y); o.y = pk2(f.z, f.w);
        ((uint2*)out)[i] = o;
    }
}

// ---------------- LayerNorm (one row / block) ----------------
__global__ __launch_bounds__(256) void ln_kernel(const float* __restrict__ x,
                                                 const float* __restrict__ w,
                                                 const float* __restrict__ b,
                                                 __bf16* __restrict__ y) {
    __shared__ float red[16];
    const int tid = threadIdx.x, lane = tid & 31, wave = tid >> 5;
    const long row = blockIdx.x;
    float4 f = *(const float4*)(x + row * Dc + tid * 4);
    float s  = f.x + f.y + f.z + f.w;
    float ss = f.x * f.x + f.y * f.y + f.z * f.z + f.w * f.w;
    #pragma unroll
    for (int off = 16; off; off >>= 1) { s += __shfl_xor(s, off, 32); ss += __shfl_xor(ss, off, 32); }
    if (lane == 0) { red[wave] = s; red[8 + wave] = ss; }
    __syncthreads();
    float ts = 0.f, tss = 0.f;
    #pragma unroll
    for (int i = 0; i < 8; ++i) { ts += red[i]; tss += red[8 + i]; }
    const float mu  = ts * (1.0f / Dc);
    const float var = tss * (1.0f / Dc) - mu * mu;
    const float rs  = rsqrtf(var + 1e-5f);
    const int c = tid * 4;
    float4 wv = *(const float4*)(w + c);
    float4 bv = *(const float4*)(b + c);
    uint2 o;
    o.x = pk2((f.x - mu) * rs * wv.x + bv.x, (f.y - mu) * rs * wv.y + bv.y);
    o.y = pk2((f.z - mu) * rs * wv.z + bv.z, (f.w - mu) * rs * wv.w + bv.w);
    *(uint2*)(y + row * Dc + c) = o;
}

// ---------------- big WMMA GEMM: C = epi(A[N,K]bf16 @ B[K,Dout]bf16 + bias) ----------------
// Double-buffered LDS, A tile staged with async global->LDS, one barrier per K step.
// EPI: 0 = f32 out (bias only), 1 = f32 out (+residual), 2 = bf16 out (exact GELU)
template <int EPI>
__global__ __launch_bounds__(256) void gemm_bf16_kernel(
    const __bf16* __restrict__ A, const __bf16* __restrict__ Bw,
    const float* __restrict__ bias, const float* __restrict__ res,
    float* __restrict__ outF, __bf16* __restrict__ outB, int K, int Dout) {
    __shared__ alignas(16) __bf16 As[2][128][40];
    __shared__ alignas(16) __bf16 Bs[2][128][40];
    const int tid = threadIdx.x, lane = tid & 31, wave = tid >> 5;
    const int wr = wave & 3, wc = wave >> 2;
    const long rowBlock = (long)blockIdx.y * 128;
    const long colBlock = (long)blockIdx.x * 128;

    f32x8 z8 = {0.f,0.f,0.f,0.f,0.f,0.f,0.f,0.f};
    f32x8 acc[2][4];
    #pragma unroll
    for (int i = 0; i < 2; ++i)
        #pragma unroll
        for (int j = 0; j < 4; ++j) acc[i][j] = z8;

    const int ar = tid >> 1, ah = (tid & 1) * 16;     // A staging: row, col half
    const int bk = tid >> 3, bc = (tid & 7) * 16;     // B staging: k, col chunk
    const int lrow = lane & 15;
    const int kA = (lane < 16) ? 0 : 8;
    const int kB = (lane < 16) ? 0 : 8;

    const __bf16* gA = A + (rowBlock + ar) * (long)K + ah;   // + k0
    const __bf16* gB = Bw + (long)bk * Dout + colBlock + bc; // + k0*Dout

    // ---- prologue: stage tile 0 into buffer 0 ----
    {
        async_b128(gA,     &As[0][ar][ah]);
        async_b128(gA + 8, &As[0][ar][ah + 8]);
        union { uint4 q[2]; __bf16 h[16]; } bu;
        bu.q[0] = ((const uint4*)gB)[0];
        bu.q[1] = ((const uint4*)gB)[1];
        #pragma unroll
        for (int j = 0; j < 16; ++j) Bs[0][bc + j][bk] = bu.h[j];
        wait_async();
        __syncthreads();
    }

    for (int k0 = 0; k0 < K; k0 += 32) {
        const int cur = (k0 >> 5) & 1, nxt = cur ^ 1;
        const bool more = (k0 + 32 < K);
        union { uint4 q[2]; __bf16 h[16]; } bu;
        if (more) {
            const __bf16* ga = gA + (k0 + 32);
            async_b128(ga,     &As[nxt][ar][ah]);
            async_b128(ga + 8, &As[nxt][ar][ah + 8]);
            const __bf16* gb = gB + (long)(k0 + 32) * Dout;
            bu.q[0] = ((const uint4*)gb)[0];
            bu.q[1] = ((const uint4*)gb)[1];
            if (k0 + 64 < K) __builtin_prefetch(gb + 32l * Dout, 0, 0);
        }

        ABFrag fa[2], fb[4];
        #pragma unroll
        for (int rt = 0; rt < 2; ++rt) {
            const __bf16* p = &As[cur][wr * 32 + rt * 16 + lrow][kA];
            fa[rt].q[0] = *(const uint4*)p;
            fa[rt].q[1] = *(const uint4*)(p + 16);
        }
        #pragma unroll
        for (int ct = 0; ct < 4; ++ct) {
            const __bf16* p = &Bs[cur][wc * 64 + ct * 16 + lrow][kB];
            fb[ct].q[0] = *(const uint4*)p;
            fb[ct].q[1] = *(const uint4*)(p + 16);
        }
        #pragma unroll
        for (int rt = 0; rt < 2; ++rt)
            #pragma unroll
            for (int ct = 0; ct < 4; ++ct)
                acc[rt][ct] = __builtin_amdgcn_wmma_f32_16x16x32_bf16(
                    false, fa[rt].v, false, fb[ct].v, (short)0, acc[rt][ct], false, false);

        if (more) {
            #pragma unroll
            for (int j = 0; j < 16; ++j) Bs[nxt][bc + j][bk] = bu.h[j];
            wait_async();
        }
        __syncthreads();
    }

    const int rowOff = (lane >= 16) ? 8 : 0;
    #pragma unroll
    for (int rt = 0; rt < 2; ++rt)
        #pragma unroll
        for (int ct = 0; ct < 4; ++ct) {
            const long col = colBlock + wc * 64 + ct * 16 + lrow;
            const float bv = bias[col];
            #pragma unroll
            for (int j = 0; j < 8; j += 2) {
                const long row0 = rowBlock + wr * 32 + rt * 16 + j + rowOff;
                float v0 = acc[rt][ct][j]     + bv;
                float v1 = acc[rt][ct][j + 1] + bv;
                if (EPI == 1) {
                    v0 += res[row0 * Dout + col];
                    v1 += res[(row0 + 1) * Dout + col];
                    outF[row0 * Dout + col]       = v0;
                    outF[(row0 + 1) * Dout + col] = v1;
                } else if (EPI == 2) {
                    float g0 = 0.5f * v0 * (1.0f + erff(v0 * 0.70710678118654752f));
                    float g1 = 0.5f * v1 * (1.0f + erff(v1 * 0.70710678118654752f));
                    __bf16* p = outB + row0 * Dout + col;
                    st2(p, p + Dout, g0, g1);
                } else {
                    outF[row0 * Dout + col]       = v0;
                    outF[(row0 + 1) * Dout + col] = v1;
                }
            }
        }
}

// ---------------- phi: out = exp(scale*data @ proj^T - |scale*data|^2/2 + eps) ----------------
__global__ __launch_bounds__(256) void phi_kernel(const float* __restrict__ src,     // [N, D]
                                                  const __bf16* __restrict__ projb,  // [M, HD]
                                                  __bf16* __restrict__ outP) {       // [B*H, L, M]
    __shared__ alignas(16) __bf16 As[128][72];
    __shared__ alignas(16) __bf16 Bs[256][72];
    __shared__ float psum[256];
    __shared__ float rowsq[128];
    const int tid = threadIdx.x, lane = tid & 31, wave = tid >> 5;
    const int bh = blockIdx.y, b = bh >> 4, h = bh & 15;
    const long l0 = (long)blockIdx.x * 128;
    const float SCL = 0.35355339059327379f;   // HD^-0.25

    {   // proj [M, HD] row-major is exactly the Bs[col][k] layout: async copy
        const __bf16* gp = projb + (long)tid * 64;
        #pragma unroll
        for (int i = 0; i < 8; ++i) async_b128(gp + i * 8, &Bs[tid][i * 8]);
    }
    const int r = tid >> 1, halfOff = (tid & 1) * 32;
    const float* gsrc = src + ((long)(b * Lc) + l0 + r) * Dc + h * 64 + halfOff;
    float s = 0.f;
    #pragma unroll
    for (int i = 0; i < 8; ++i) {
        float4 f = ((const float4*)gsrc)[i];
        f.x *= SCL; f.y *= SCL; f.z *= SCL; f.w *= SCL;
        s += f.x * f.x + f.y * f.y + f.z * f.z + f.w * f.w;
        *(unsigned*)&As[r][halfOff + i * 4 + 0] = pk2(f.x, f.y);
        *(unsigned*)&As[r][halfOff + i * 4 + 2] = pk2(f.z, f.w);
    }
    psum[tid] = s;
    wait_async();
    __syncthreads();
    if (tid < 128) rowsq[tid] = 0.5f * (psum[2 * tid] + psum[2 * tid + 1]);
    __syncthreads();

    const int lrow = lane & 15;
    const int kA = (lane < 16) ? 0 : 8;
    const int kB = (lane < 16) ? 0 : 8;
    const int rowOff = (lane >= 16) ? 8 : 0;

    ABFrag fa[2];
    #pragma unroll
    for (int ks = 0; ks < 2; ++ks) {
        const __bf16* p = &As[wave * 16 + lrow][ks * 32 + kA];
        fa[ks].q[0] = *(const uint4*)p;
        fa[ks].q[1] = *(const uint4*)(p + 16);
    }
    // four independent accumulator chains hide the BF16 WMMA->WMMA RAW hazard (5 slots)
    for (int ct = 0; ct < 16; ct += 4) {
        f32x8 accs[4];
        #pragma unroll
        for (int t = 0; t < 4; ++t) accs[t] = (f32x8){0.f,0.f,0.f,0.f,0.f,0.f,0.f,0.f};
        #pragma unroll
        for (int ks = 0; ks < 2; ++ks) {
            ABFrag fb[4];
            #pragma unroll
            for (int t = 0; t < 4; ++t) {
                const __bf16* p = &Bs[(ct + t) * 16 + lrow][ks * 32 + kB];
                fb[t].q[0] = *(const uint4*)p;
                fb[t].q[1] = *(const uint4*)(p + 16);
            }
            #pragma unroll
            for (int t = 0; t < 4; ++t)
                accs[t] = __builtin_amdgcn_wmma_f32_16x16x32_bf16(false, fa[ks].v, false, fb[t].v,
                                                                 (short)0, accs[t], false, false);
        }
        #pragma unroll
        for (int t = 0; t < 4; ++t) {
            const long col = (ct + t) * 16 + lrow;
            #pragma unroll
            for (int j = 0; j < 8; j += 2) {
                const int row0 = wave * 16 + j + rowOff;
                float e0 = __expf(accs[t][j]     - rowsq[row0]     + 1e-6f);
                float e1 = __expf(accs[t][j + 1] - rowsq[row0 + 1] + 1e-6f);
                __bf16* p = outP + ((long)bh * Lc + l0 + row0) * Mc + col;
                st2(p, p + Mc, e0, e1);
            }
        }
    }
}

// ---------------- kv = k_p^T @ v  (+ fused k_sum) ----------------
__global__ __launch_bounds__(256) void kv_ksum_kernel(const __bf16* __restrict__ kp, // [B*H,L,M]
                                                      const float* __restrict__ v,   // [N,D]
                                                      __bf16* __restrict__ kv,       // [B*H,M,HD]
                                                      float* __restrict__ ksum) {    // [B*H,M]
    __shared__ alignas(16) __bf16 As[256][40];
    __shared__ alignas(16) __bf16 Bs[64][40];
    const int tid = threadIdx.x, lane = tid & 31, wave = tid >> 5;
    const int bh = blockIdx.x, b = bh >> 4, h = bh & 15;

    f32x8 z8 = {0.f,0.f,0.f,0.f,0.f,0.f,0.f,0.f};
    f32x8 acc[2][4];
    #pragma unroll
    for (int i = 0; i < 2; ++i)
        #pragma unroll
        for (int j = 0; j < 4; ++j) acc[i][j] = z8;
    float ks_acc = 0.f;

    const int al = tid >> 3, am = (tid & 7) * 32;   // A staging: l local, m base
    const int bl = tid & 31, bd = (tid >> 5) * 8;   // B staging: l local, d base
    const int lrow = lane & 15;
    const int kA = (lane < 16) ? 0 : 8;
    const int kB = (lane < 16) ? 0 : 8;

    for (int l0 = 0; l0 < Lc; l0 += 32) {
        const __bf16* gk = kp + ((long)bh * Lc + l0 + al) * Mc + am;
        union { uint4 q[4]; __bf16 h16[32]; } au;
        au.q[0] = ((const uint4*)gk)[0]; au.q[1] = ((const uint4*)gk)[1];
        au.q[2] = ((const uint4*)gk)[2]; au.q[3] = ((const uint4*)gk)[3];
        #pragma unroll
        for (int j = 0; j < 32; ++j) As[am + j][al] = au.h16[j];

        const float* gv = v + ((long)(b * Lc) + l0 + bl) * Dc + h * 64 + bd;
        float4 f0 = ((const float4*)gv)[0];
        float4 f1 = ((const float4*)gv)[1];
        Bs[bd + 0][bl] = f2bf(f0.x); Bs[bd + 1][bl] = f2bf(f0.y);
        Bs[bd + 2][bl] = f2bf(f0.z); Bs[bd + 3][bl] = f2bf(f0.w);
        Bs[bd + 4][bl] = f2bf(f1.x); Bs[bd + 5][bl] = f2bf(f1.y);
        Bs[bd + 6][bl] = f2bf(f1.z); Bs[bd + 7][bl] = f2bf(f1.w);
        __syncthreads();

        #pragma unroll
        for (int c = 0; c < 4; ++c) {     // thread tid owns m=tid; vectorized row-sum
            union { uint4 q; __bf16 h[8]; } u;
            u.q = *(const uint4*)&As[tid][c * 8];
            #pragma unroll
            for (int j = 0; j < 8; ++j) ks_acc += bf2f(u.h[j]);
        }

        ABFrag fa[2], fb[4];
        #pragma unroll
        for (int rt = 0; rt < 2; ++rt) {
            const __bf16* p = &As[wave * 32 + rt * 16 + lrow][kA];
            fa[rt].q[0] = *(const uint4*)p;
            fa[rt].q[1] = *(const uint4*)(p + 16);
        }
        #pragma unroll
        for (int ct = 0; ct < 4; ++ct) {
            const __bf16* p = &Bs[ct * 16 + lrow][kB];
            fb[ct].q[0] = *(const uint4*)p;
            fb[ct].q[1] = *(const uint4*)(p + 16);
        }
        #pragma unroll
        for (int rt = 0; rt < 2; ++rt)
            #pragma unroll
            for (int ct = 0; ct < 4; ++ct)
                acc[rt][ct] = __builtin_amdgcn_wmma_f32_16x16x32_bf16(
                    false, fa[rt].v, false, fb[ct].v, (short)0, acc[rt][ct], false, false);
        __syncthreads();
    }

    ksum[(long)bh * Mc + tid] = ks_acc;
    const int rowOff = (lane >= 16) ? 8 : 0;
    #pragma unroll
    for (int rt = 0; rt < 2; ++rt)
        #pragma unroll
        for (int ct = 0; ct < 4; ++ct)
            #pragma unroll
            for (int j = 0; j < 8; j += 2) {
                const int m = wave * 32 + rt * 16 + j + rowOff;
                const int d = ct * 16 + lrow;
                __bf16* p = kv + ((long)bh * Mc + m) * HDc + d;
                st2(p, p + HDc, acc[rt][ct][j], acc[rt][ct][j + 1]);
            }
}

// ---------------- z = 1 / (q_p . k_sum + 1e-6) ----------------
__global__ __launch_bounds__(256) void z_kernel(const __bf16* __restrict__ qp,
                                                const float* __restrict__ ksum,
                                                float* __restrict__ z) {
    const int lane = threadIdx.x & 31, wave = threadIdx.x >> 5;
    const long row = (long)blockIdx.x * 8 + wave;        // [0, B*H*L)
    const int bh = (int)(row >> 12);                     // L = 4096
    union { uint4 q; __bf16 h[8]; } qu;
    qu.q = *(const uint4*)(qp + row * Mc + lane * 8);
    const float* gs = ksum + (long)bh * Mc + lane * 8;
    float4 s0 = ((const float4*)gs)[0], s1 = ((const float4*)gs)[1];
    float d = bf2f(qu.h[0]) * s0.x + bf2f(qu.h[1]) * s0.y +
              bf2f(qu.h[2]) * s0.z + bf2f(qu.h[3]) * s0.w +
              bf2f(qu.h[4]) * s1.x + bf2f(qu.h[5]) * s1.y +
              bf2f(qu.h[6]) * s1.z + bf2f(qu.h[7]) * s1.w;
    #pragma unroll
    for (int off = 16; off; off >>= 1) d += __shfl_xor(d, off, 32);
    if (lane == 0) z[row] = 1.0f / (d + 1e-6f);
}

// ---------------- attn = (q_p @ kv) * z, written head-interleaved bf16 [N,D] ----------------
__global__ __launch_bounds__(256) void attn_kernel(const __bf16* __restrict__ qp,  // [B*H,L,M]
                                                   const __bf16* __restrict__ kv,  // [B*H,M,HD]
                                                   const float* __restrict__ z,    // [B*H,L]
                                                   __bf16* __restrict__ attn) {    // [N,D]
    __shared__ alignas(16) __bf16 As[128][40];
    __shared__ alignas(16) __bf16 Bs[64][40];
    __shared__ float zs[128];
    const int tid = threadIdx.x, lane = tid & 31, wave = tid >> 5;
    const int bh = blockIdx.y, b = bh >> 4, h = bh & 15;
    const long l0 = (long)blockIdx.x * 128;
    if (tid < 128) zs[tid] = z[(long)bh * Lc + l0 + tid];

    f32x8 acc[4];
    f32x8 z8 = {0.f,0.f,0.f,0.f,0.f,0.f,0.f,0.f};
    #pragma unroll
    for (int j = 0; j < 4; ++j) acc[j] = z8;

    const int ar = tid >> 1, ah = (tid & 1) * 16;
    const int bm = tid & 31, bd = (tid >> 5) * 8;
    const int lrow = lane & 15;
    const int kA = (lane < 16) ? 0 : 8;
    const int kB = (lane < 16) ? 0 : 8;

    for (int m0 = 0; m0 < Mc; m0 += 32) {
#if __has_builtin(__builtin_amdgcn_tensor_load_to_lds)
        // TDM: [128 x 32] bf16 tile of q_p (row stride Mc), rows padded into As[128][40]
        if (wave == 0) {
            const __bf16* gt = qp + ((long)bh * Lc + l0) * Mc + m0;
            tdm_load_tile_2d(gt, (unsigned)(size_t)&As[0][0], 32u, 128u,
                             (unsigned)Mc, (unsigned)Lc, (unsigned long long)Mc);
            __builtin_amdgcn_s_wait_tensorcnt(0);
        }
#else
        const __bf16* gq = qp + ((long)bh * Lc + l0 + ar) * Mc + m0 + ah;
        *(uint4*)&As[ar][ah]     = ((const uint4*)gq)[0];
        *(uint4*)&As[ar][ah + 8] = ((const uint4*)gq)[1];
#endif
        union { uint4 q; __bf16 h8[8]; } ku;
        ku.q = *(const uint4*)(kv + ((long)bh * Mc + m0 + bm) * HDc + bd);
        #pragma unroll
        for (int j = 0; j < 8; ++j) Bs[bd + j][bm] = ku.h8[j];
        __syncthreads();

        ABFrag fa;
        {
            const __bf16* p = &As[wave * 16 + lrow][kA];
            fa.q[0] = *(const uint4*)p;
            fa.q[1] = *(const uint4*)(p + 16);
        }
        #pragma unroll
        for (int ct = 0; ct < 4; ++ct) {
            ABFrag fb;
            const __bf16* p = &Bs[ct * 16 + lrow][kB];
            fb.q[0] = *(const uint4*)p;
            fb.q[1] = *(const uint4*)(p + 16);
            acc[ct] = __builtin_amdgcn_wmma_f32_16x16x32_bf16(false, fa.v, false, fb.v,
                                                              (short)0, acc[ct], false, false);
        }
        __syncthreads();
    }

    const int rowOff = (lane >= 16) ? 8 : 0;
    #pragma unroll
    for (int ct = 0; ct < 4; ++ct)
        #pragma unroll
        for (int j = 0; j < 8; j += 2) {
            const int row0 = wave * 16 + j + rowOff;
            const int col = ct * 16 + lrow;
            float o0 = acc[ct][j]     * zs[row0];
            float o1 = acc[ct][j + 1] * zs[row0 + 1];
            __bf16* p = attn + ((long)(b * Lc) + l0 + row0) * Dc + h * 64 + col;
            st2(p, p + Dc, o0, o1);
        }
}

// ---------------- host orchestration ----------------
extern "C" void kernel_launch(void* const* d_in, const int* in_sizes, int n_in,
                              void* d_out, int out_size, void* d_ws, size_t ws_size,
                              hipStream_t stream) {
    (void)in_sizes; (void)n_in; (void)out_size; (void)ws_size;
    const float* x    = (const float*)d_in[0];
    const float* ln1w = (const float*)d_in[1];
    const float* ln1b = (const float*)d_in[2];
    const float* ln2w = (const float*)d_in[3];
    const float* ln2b = (const float*)d_in[4];
    const float* Wq = (const float*)d_in[5];  const float* bq = (const float*)d_in[6];
    const float* Wk = (const float*)d_in[7];  const float* bk = (const float*)d_in[8];
    const float* Wv = (const float*)d_in[9];  const float* bv = (const float*)d_in[10];
    const float* Wo = (const float*)d_in[11]; const float* bo = (const float*)d_in[12];
    const float* W1 = (const float*)d_in[13]; const float* b1 = (const float*)d_in[14];
    const float* W2 = (const float*)d_in[15]; const float* b2 = (const float*)d_in[16];
    const float* proj = (const float*)d_in[17];
    float* outp = (float*)d_out;

    char* ws = (char*)d_ws;
    constexpr size_t SZ_W    = (size_t)Dc * Dc * 2;
    constexpr size_t SZ_W14  = (size_t)Dc * D4 * 2;
    constexpr size_t SZ_PROJ = (size_t)Mc * HDc * 2;
    constexpr size_t SZ_H    = (size_t)Nc * Dc * 2;
    constexpr size_t SZ_F32  = (size_t)Nc * Dc * 4;
    constexpr size_t SZ_P    = (size_t)Bc * Hc * Lc * Mc * 2;
    constexpr size_t SZ_KV   = (size_t)Bc * Hc * Mc * HDc * 2;
    constexpr size_t SZ_KSUM = (size_t)Bc * Hc * Mc * 4;
    constexpr size_t SZ_Z    = (size_t)Bc * Hc * Lc * 4;

    size_t o = 0;
    const size_t o_wq   = o; o += SZ_W;
    const size_t o_wk   = o; o += SZ_W;
    const size_t o_wv   = o; o += SZ_W;
    const size_t o_wo   = o; o += SZ_W;
    const size_t o_w1   = o; o += SZ_W14;
    const size_t o_w2   = o; o += SZ_W14;
    const size_t o_proj = o; o += SZ_PROJ;
    const size_t o_h    = o; o += SZ_H;       // h / attn_bf16 / h2 (sequential reuse)
    const size_t o_q    = o; o += SZ_F32;     // q f32; region q..v reused by G bf16
    const size_t o_k    = o; o += SZ_F32;
    const size_t o_v    = o; o += SZ_F32;
    const size_t o_p    = o; o += SZ_P;       // k_p then q_p
    const size_t o_kv   = o; o += SZ_KV;
    const size_t o_ks   = o; o += SZ_KSUM;
    const size_t o_z    = o; o += SZ_Z;
    const size_t o_x2   = o; o += SZ_F32;

    __bf16* wqb = (__bf16*)(ws + o_wq); __bf16* wkb = (__bf16*)(ws + o_wk);
    __bf16* wvb = (__bf16*)(ws + o_wv); __bf16* wob = (__bf16*)(ws + o_wo);
    __bf16* w1b = (__bf16*)(ws + o_w1); __bf16* w2b = (__bf16*)(ws + o_w2);
    __bf16* projb = (__bf16*)(ws + o_proj);
    __bf16* hbuf  = (__bf16*)(ws + o_h);
    float*  qf = (float*)(ws + o_q);
    float*  kf = (float*)(ws + o_k);
    float*  vf = (float*)(ws + o_v);
    __bf16* Gbuf  = (__bf16*)(ws + o_q);      // aliases dead q/k/v region (128MB <= 192MB)
    __bf16* Pbuf  = (__bf16*)(ws + o_p);
    __bf16* kvbuf = (__bf16*)(ws + o_kv);
    float*  ksbuf = (float*)(ws + o_ks);
    float*  zbuf  = (float*)(ws + o_z);
    float*  x2    = (float*)(ws + o_x2);

    // 1) weight conversions (every launch: deterministic, no caching)
    auto cvt = [&](const float* in, __bf16* out, long n) {
        long n4 = n / 4;
        cvt_f32_bf16<<<(unsigned)((n4 + 255) / 256), 256, 0, stream>>>(in, out, n4);
    };
    cvt(Wq, wqb, (long)Dc * Dc); cvt(Wk, wkb, (long)Dc * Dc);
    cvt(Wv, wvb, (long)Dc * Dc); cvt(Wo, wob, (long)Dc * Dc);
    cvt(W1, w1b, (long)Dc * D4); cvt(W2, w2b, (long)Dc * D4);
    cvt(proj, projb, (long)Mc * HDc);

    // 2) LN1 -> h
    ln_kernel<<<Nc, 256, 0, stream>>>(x, ln1w, ln1b, hbuf);

    // 3) Q/K/V projections
    gemm_bf16_kernel<0><<<dim3(Dc / 128, Nc / 128), 256, 0, stream>>>(hbuf, wqb, bq, nullptr, qf, nullptr, Dc, Dc);
    gemm_bf16_kernel<0><<<dim3(Dc / 128, Nc / 128), 256, 0, stream>>>(hbuf, wkb, bk, nullptr, kf, nullptr, Dc, Dc);
    gemm_bf16_kernel<0><<<dim3(Dc / 128, Nc / 128), 256, 0, stream>>>(hbuf, wvb, bv, nullptr, vf, nullptr, Dc, Dc);

    // 4) k_p, then kv + k_sum, then q_p (reusing the same buffer), z
    phi_kernel<<<dim3(Lc / 128, Bc * Hc), 256, 0, stream>>>(kf, projb, Pbuf);
    kv_ksum_kernel<<<Bc * Hc, 256, 0, stream>>>(Pbuf, vf, kvbuf, ksbuf);
    phi_kernel<<<dim3(Lc / 128, Bc * Hc), 256, 0, stream>>>(qf, projb, Pbuf);
    z_kernel<<<(Bc * Hc * Lc) / 8, 256, 0, stream>>>(Pbuf, ksbuf, zbuf);

    // 5) attn (bf16, head-interleaved into hbuf), then output projection + residual
    attn_kernel<<<dim3(Lc / 128, Bc * Hc), 256, 0, stream>>>(Pbuf, kvbuf, zbuf, hbuf);
    gemm_bf16_kernel<1><<<dim3(Dc / 128, Nc / 128), 256, 0, stream>>>(hbuf, wob, bo, x, x2, nullptr, Dc, Dc);

    // 6) LN2 -> h2 (hbuf reuse), MLP
    ln_kernel<<<Nc, 256, 0, stream>>>(x2, ln2w, ln2b, hbuf);
    gemm_bf16_kernel<2><<<dim3(D4 / 128, Nc / 128), 256, 0, stream>>>(hbuf, w1b, b1, nullptr, nullptr, Gbuf, Dc, D4);
    gemm_bf16_kernel<1><<<dim3(Dc / 128, Nc / 128), 256, 0, stream>>>(Gbuf, w2b, b2, x2, outp, nullptr, D4, Dc);
}
